// Model_53197464928391
// MI455X (gfx1250) — matrix operand order repaired and checked by reading, not compiled, not run on hardware
//
#include <hip/hip_runtime.h>
#include <hip/hip_bf16.h>
#include <math.h>

// ---------------- problem constants (match reference) ----------------
#define N_NODES 10000
#define FEAT    2048
#define RELS    8
#define NE      160000
#define NT      100000
#define DCAT    896          // 512 + 256 + 128
#define BN_RSQ  0.9999950000374997f   // 1/sqrt(1 + 1e-5)

typedef __bf16 bf16_t;
typedef __attribute__((ext_vector_type(16))) __bf16 v16bf;
typedef __attribute__((ext_vector_type(8)))  __bf16 v8bf;
typedef __attribute__((ext_vector_type(8)))  float  v8f;

static __device__ __forceinline__ bf16_t f2bf(float f) {
  union { float f; unsigned u; } a; a.f = f;
  unsigned r = a.u + 0x7FFFu + ((a.u >> 16) & 1u);   // round-to-nearest-even
  unsigned short h = (unsigned short)(r >> 16);
  return __builtin_bit_cast(bf16_t, h);
}

// ---------------- f32 -> bf16 conversion ----------------
__global__ void k_f32_to_bf16(const float* __restrict__ src, bf16_t* __restrict__ dst, size_t n) {
  size_t i = (size_t)blockIdx.x * (size_t)blockDim.x * 4 + threadIdx.x;
  #pragma unroll
  for (int j = 0; j < 4; ++j) {
    size_t idx = i + (size_t)j * blockDim.x;
    if (idx < n) dst[idx] = f2bf(src[idx]);
  }
}

// ---------------- weight transpose + convert: w (K x N, f32) -> wt (N x K, bf16) ----------------
__global__ void k_w_transpose_bf16(const float* __restrict__ w, bf16_t* __restrict__ wt,
                                   int K, int N) {
  size_t i = (size_t)blockIdx.x * blockDim.x + threadIdx.x;
  if (i >= (size_t)K * N) return;
  int n = (int)(i / (size_t)K);
  int k = (int)(i % (size_t)K);
  wt[i] = f2bf(w[(size_t)k * N + n]);
}

// ---------------- tiled bf16 WMMA GEMM with fused epilogue ----------------
// A: M x K row-major bf16 (lda). Bt: N x K row-major bf16 (= B transposed, ldbt).
// C = act(acc + bias [+ C_old]) [* bn_g' + bn_b] ; dual fp32 / bf16 output.
// act: 0=none 1=elu 2=relu 3=sigmoid
__global__ __launch_bounds__(256)
void k_gemm_bf16_wmma(const bf16_t* __restrict__ A, int lda,
                      const bf16_t* __restrict__ Bt, int ldbt,
                      int M, int N, int K,
                      const float* __restrict__ bias,
                      const float* __restrict__ bng, const float* __restrict__ bnb,
                      int act, int accum,
                      float* __restrict__ Cf, int ldcf,
                      bf16_t* __restrict__ Cb, int ldcb)
{
  constexpr int BM = 128, BN = 128, BK = 32;
  __shared__ bf16_t As[BM * BK];      // [m][k]
  __shared__ bf16_t Bs[BN * BK];      // [n][k]  (Bt is already n-major)

  const int tid  = threadIdx.x;
  const int lane = tid & 31;
  const int wave = tid >> 5;          // 8 waves
  const int wrow = wave >> 1;         // 0..3  -> 32-row strip
  const int wcol = wave & 1;          // 0..1  -> 64-col strip
  const int l15  = lane & 15;
  const int lhi  = lane >> 4;

  const int tile_m = blockIdx.y * BM;
  const int tile_n = blockIdx.x * BN;

  v8f acc[2][4] = {};

  const int st_row = tid >> 1;               // 0..127 (used for both A-row and Bt-row)
  const int st_kc  = (tid & 1) * 16;         // 0 / 16

  for (int k0 = 0; k0 < K; k0 += BK) {
    // ---- stage A tile (K is always a multiple of 32 in this model) ----
    {
      int gr = tile_m + st_row;
      bf16_t* dst = &As[st_row * BK + st_kc];
      if (gr < M) {
        const bf16_t* src = A + (size_t)gr * lda + (k0 + st_kc);
        *(v8bf*)(dst)     = *(const v8bf*)(src);
        *(v8bf*)(dst + 8) = *(const v8bf*)(src + 8);
      } else {
        v8bf z = {};
        *(v8bf*)(dst) = z; *(v8bf*)(dst + 8) = z;
      }
    }
    // ---- stage Bt tile: contiguous vector copies, no scalar transpose ----
    {
      int gc = tile_n + st_row;
      bf16_t* dst = &Bs[st_row * BK + st_kc];
      if (gc < N) {
        const bf16_t* src = Bt + (size_t)gc * ldbt + (k0 + st_kc);
        *(v8bf*)(dst)     = *(const v8bf*)(src);
        *(v8bf*)(dst + 8) = *(const v8bf*)(src + 8);
      } else {
        v8bf z = {};
        *(v8bf*)(dst) = z; *(v8bf*)(dst + 8) = z;
      }
    }
    __syncthreads();

    if (k0 + BK < K) {   // global_prefetch_b8 of next tiles
      __builtin_prefetch(A  + (size_t)(tile_m + st_row) * lda  + (k0 + BK + st_kc), 0, 1);
      __builtin_prefetch(Bt + (size_t)(tile_n + st_row) * ldbt + (k0 + BK + st_kc), 0, 1);
    }

    union F16 { v16bf v; v8bf h[2]; };
    F16 af[2]; F16 bfr[4];
    #pragma unroll
    for (int tm = 0; tm < 2; ++tm) {
      int r  = wrow * 32 + tm * 16 + l15;
      int kb = lhi * 8;                         // lanes>=16 take K 8..15 / 24..31
      af[tm].h[0] = *(const v8bf*)&As[r * BK + kb];
      af[tm].h[1] = *(const v8bf*)&As[r * BK + 16 + kb];
    }
    #pragma unroll
    for (int tn = 0; tn < 4; ++tn) {
      int c  = wcol * 64 + tn * 16 + l15;
      int kh = lhi * 8;                         // lanes>=16 take K 8..15 / 24..31
      bfr[tn].h[0] = *(const v8bf*)&Bs[c * BK + kh];
      bfr[tn].h[1] = *(const v8bf*)&Bs[c * BK + 16 + kh];
    }
    #pragma unroll
    for (int tm = 0; tm < 2; ++tm)
      #pragma unroll
      for (int tn = 0; tn < 4; ++tn)
        acc[tm][tn] = __builtin_amdgcn_wmma_f32_16x16x32_bf16(
            false, af[tm].v, false, bfr[tn].v, (short)0, acc[tm][tn], false, false);
    __syncthreads();
  }

  // ---- fused epilogue: (+bias [+C_old]) -> act -> BN -> f32/bf16 stores ----
  #pragma unroll
  for (int tn = 0; tn < 4; ++tn) {
    int col = tile_n + wcol * 64 + tn * 16 + l15;
    if (col >= N) continue;
    float bv = bias ? bias[col] : 0.f;
    float g  = bng ? bng[col] * BN_RSQ : 1.f;
    float sh = bnb ? bnb[col] : 0.f;
    #pragma unroll
    for (int tm = 0; tm < 2; ++tm) {
      #pragma unroll
      for (int v = 0; v < 8; ++v) {
        int row = tile_m + wrow * 32 + tm * 16 + lhi * 8 + v;
        if (row >= M) continue;
        float x = acc[tm][tn][v] + bv;
        if (accum && Cf) x += Cf[(size_t)row * ldcf + col];
        if      (act == 1) x = (x > 0.f) ? x : (expf(x) - 1.f);
        else if (act == 2) x = fmaxf(x, 0.f);
        else if (act == 3) x = 1.f / (1.f + expf(-x));
        if (bng) x = x * g + sh;
        if (Cf) Cf[(size_t)row * ldcf + col] = x;
        if (Cb) Cb[(size_t)row * ldcb + col] = f2bf(x);
      }
    }
  }
}

// ---------------- segment-mean building blocks ----------------
__global__ void k_scatter_feats(const float* __restrict__ x, int ldx,
                                const int* __restrict__ src, const int* __restrict__ dst,
                                const int* __restrict__ etype, int rels,
                                float* __restrict__ agg, int F)
{
  int e = blockIdx.x;
  int s = src[e];
  int seg = dst[e] * rels + (etype ? etype[e] : 0);
  const float* xr = x + (size_t)s * ldx;
  float* ar = agg + (size_t)seg * F;
  for (int f = threadIdx.x; f < F; f += blockDim.x)
    atomicAdd(&ar[f], xr[f]);                       // global_atomic_add_f32
}

__global__ void k_scatter_counts(const int* __restrict__ dst, const int* __restrict__ etype,
                                 int rels, float* __restrict__ cnt, int E)
{
  int e = blockIdx.x * blockDim.x + threadIdx.x;
  if (e >= E) return;
  int seg = dst[e] * rels + (etype ? etype[e] : 0);
  atomicAdd(&cnt[seg], 1.f);
}

__global__ void k_seg_div_bf16(const float* __restrict__ agg, const float* __restrict__ cnt,
                               bf16_t* __restrict__ out, size_t total, int F)
{
  size_t i = (size_t)blockIdx.x * blockDim.x + threadIdx.x;
  if (i >= total) return;
  size_t seg = i / (size_t)F;
  float c = fmaxf(cnt[seg], 1.f);
  out[i] = f2bf(agg[i] / c);
}

// ---------------- attention over 3 views ----------------
__global__ void k_att_scores(const float* __restrict__ ae, const float* __restrict__ dg,
                             const float* __restrict__ og,
                             const float* __restrict__ w1, const float* __restrict__ b1,
                             const float* __restrict__ w2, float* __restrict__ beta)
{
  int n = blockIdx.x * blockDim.x + threadIdx.x;
  if (n >= N_NODES) return;
  const float* zs[3] = { ae + (size_t)n * DCAT, dg + (size_t)n * DCAT, og + (size_t)n * DCAT };
  float s[3];
  #pragma unroll
  for (int v = 0; v < 3; ++v) {
    float h[8];
    #pragma unroll
    for (int j = 0; j < 8; ++j) h[j] = b1[j];
    for (int f = 0; f < DCAT; ++f) {
      float z = zs[v][f];
      #pragma unroll
      for (int j = 0; j < 8; ++j) h[j] += z * w1[f * 8 + j];
    }
    float sc = 0.f;
    #pragma unroll
    for (int j = 0; j < 8; ++j) sc += tanhf(h[j]) * w2[j];
    s[v] = sc;
  }
  float m = fmaxf(s[0], fmaxf(s[1], s[2]));
  float e0 = expf(s[0] - m), e1 = expf(s[1] - m), e2 = expf(s[2] - m);
  float inv = 1.f / (e0 + e1 + e2);
  beta[n * 3 + 0] = e0 * inv; beta[n * 3 + 1] = e1 * inv; beta[n * 3 + 2] = e2 * inv;
}

__global__ void k_att_combine(const float* __restrict__ ae, const float* __restrict__ dg,
                              const float* __restrict__ og, const float* __restrict__ beta,
                              float* __restrict__ emb)
{
  size_t i = (size_t)blockIdx.x * blockDim.x + threadIdx.x;
  if (i >= (size_t)N_NODES * DCAT) return;
  size_t n = i / DCAT;
  emb[i] = beta[n*3+0]*ae[i] + beta[n*3+1]*dg[i] + beta[n*3+2]*og[i];
}

// ---------------- triple-pair gather ----------------
__global__ void k_pair(const float* __restrict__ emb, const int* __restrict__ triples,
                       bf16_t* __restrict__ out)
{
  int t = blockIdx.x;
  int h = triples[2*t], tl = triples[2*t+1];
  const float* eh = emb + (size_t)h  * DCAT;
  const float* et = emb + (size_t)tl * DCAT;
  bf16_t* o = out + (size_t)t * DCAT;
  for (int f = threadIdx.x; f < DCAT; f += blockDim.x)
    o[f] = f2bf(0.5f * (eh[f] + et[f]));
}

// ======================================================================
extern "C" void kernel_launch(void* const* d_in, const int* in_sizes, int n_in,
                              void* d_out, int out_size, void* d_ws, size_t ws_size,
                              hipStream_t stream)
{
  (void)in_sizes; (void)n_in; (void)out_size; (void)ws_size;
  const int*   edge_index = (const int*)d_in[0];
  const int*   diff_ei    = (const int*)d_in[1];
  const int*   etype      = (const int*)d_in[2];
  const float* attr       = (const float*)d_in[3];
  const int*   triples    = (const int*)d_in[4];
  auto P = [&](int i){ return (const float*)d_in[i]; };
  const float *ae1_w=P(5),  *ae1_b=P(6),  *ae1_g=P(7),  *ae1_be=P(8);
  const float *ae2_w=P(9),  *ae2_b=P(10), *ae2_g=P(11), *ae2_be=P(12);
  const float *ae3_w=P(13), *ae3_b=P(14), *ae3_g=P(15), *ae3_be=P(16);
  const float *dw1=P(17), *db1=P(18), *dw2=P(19), *db2=P(20), *dw3=P(21), *db3=P(22);
  const float *s_wl[3]={P(23),P(26),P(29)}, *s_bl[3]={P(24),P(27),P(30)}, *s_wr[3]={P(25),P(28),P(31)};
  const float *r_wrel[3]={P(32),P(35),P(38)}, *r_root[3]={P(33),P(36),P(39)}, *r_bias[3]={P(34),P(37),P(40)};
  const float *att_w1=P(41), *att_b1=P(42), *att_w2=P(43);
  const float *d1_w=P(44), *d1_b=P(45), *d1_g=P(46), *d1_be=P(47);
  const float *d2_w=P(48), *d2_b=P(49), *d2_g=P(50), *d2_be=P(51);
  const float *d3_w=P(52), *d3_b=P(53);

  float* out_fc = (float*)d_out;                        // (T, 8)
  float* re_out = out_fc + (size_t)NT * 8;              // (N, 2048)

  char* ws = (char*)d_ws; size_t off = 0;
  auto carve = [&](size_t bytes) -> void* {
    off = (off + 255) & ~(size_t)255;
    void* p = ws + off; off += bytes; return p;
  };
  auto conv = [&](const float* s, bf16_t* d, size_t n){
    int thr = 256; long long blk = ((long long)n + thr*4 - 1) / (thr*4);
    k_f32_to_bf16<<<(int)blk, thr, 0, stream>>>(s, d, n);
  };
  // weight (K x N f32) -> transposed bf16 (N x K)
  auto mkwt = [&](const float* w, int K, int N){
    bf16_t* d = (bf16_t*)carve((size_t)K * N * 2);
    size_t tot = (size_t)K * N;
    k_w_transpose_bf16<<<(int)((tot + 255) / 256), 256, 0, stream>>>(w, d, K, N);
    return d;
  };
  auto gemm = [&](const bf16_t* A, int lda, const bf16_t* Bt, int M, int N, int K,
                  const float* bias, const float* g, const float* b, int act, int acc,
                  float* Cf, int ldcf, bf16_t* Cb, int ldcb){
    dim3 grid((N + 127) / 128, (M + 127) / 128);
    k_gemm_bf16_wmma<<<grid, 256, 0, stream>>>(A, lda, Bt, K, M, N, K,
                                               bias, g, b, act, acc, Cf, ldcf, Cb, ldcb);
  };

  // ---- bf16 inputs & transposed bf16 weights ----
  bf16_t* attr_bf = (bf16_t*)carve((size_t)N_NODES * FEAT * 2);
  conv(attr, attr_bf, (size_t)N_NODES * FEAT);
  bf16_t* ae1_wt = mkwt(ae1_w, FEAT, 512);
  bf16_t* ae2_wt = mkwt(ae2_w, 512, 256);
  bf16_t* ae3_wt = mkwt(ae3_w, 256, 128);
  bf16_t* dw1t   = mkwt(dw1, 128, 256);
  bf16_t* dw2t   = mkwt(dw2, 256, 512);
  bf16_t* dw3t   = mkwt(dw3, 512, 2048);
  int fin_[3]  = {DCAT, 512, 256};
  int fout_[3] = {512, 256, 128};
  bf16_t *s_wlt[3], *s_wrt[3], *r_wrelt[3], *r_roott[3];
  for (int l = 0; l < 3; ++l) {
    s_wlt[l]   = mkwt(s_wl[l],  fin_[l], fout_[l]);
    s_wrt[l]   = mkwt(s_wr[l],  fin_[l], fout_[l]);
    r_wrelt[l] = mkwt(r_wrel[l], RELS * fin_[l], fout_[l]);   // flat (R*fin) x fout
    r_roott[l] = mkwt(r_root[l], fin_[l], fout_[l]);
  }
  bf16_t* d1_wt = mkwt(d1_w, DCAT, 512);
  bf16_t* d2_wt = mkwt(d2_w, 512, 256);
  bf16_t* d3_wt = mkwt(d3_w, 256, 8);

  // ---- activations / scratch ----
  float*  ae_out = (float*)carve((size_t)N_NODES * DCAT * 4);
  bf16_t* ae_bf  = (bf16_t*)carve((size_t)N_NODES * DCAT * 2);
  bf16_t* dec1   = (bf16_t*)carve((size_t)N_NODES * 256 * 2);
  bf16_t* dec2   = (bf16_t*)carve((size_t)N_NODES * 512 * 2);
  float*  dg     = (float*)carve((size_t)N_NODES * DCAT * 4);
  bf16_t* dg_bf  = (bf16_t*)carve((size_t)N_NODES * DCAT * 2);
  float*  og     = (float*)carve((size_t)N_NODES * DCAT * 4);
  bf16_t* og_bf  = (bf16_t*)carve((size_t)N_NODES * DCAT * 2);
  float*  agg    = (float*)carve((size_t)N_NODES * DCAT * 4);
  bf16_t* agg_bf = (bf16_t*)carve((size_t)N_NODES * DCAT * 2);
  float*  cnt    = (float*)carve((size_t)N_NODES * 4);
  float*  aggr   = (float*)carve((size_t)N_NODES * RELS * DCAT * 4);
  bf16_t* aggr_bf= (bf16_t*)carve((size_t)N_NODES * RELS * DCAT * 2);
  float*  cntr   = (float*)carve((size_t)N_NODES * RELS * 4);
  float*  beta   = (float*)carve((size_t)N_NODES * 3 * 4);
  float*  emb    = (float*)carve((size_t)N_NODES * DCAT * 4);
  bf16_t* pairb  = (bf16_t*)carve((size_t)NT * DCAT * 2);
  bf16_t* h1     = (bf16_t*)carve((size_t)NT * 512 * 2);
  bf16_t* h2     = (bf16_t*)carve((size_t)NT * 256 * 2);

  // ---- autoencoder (z1|z2|z3 packed into ae_out, ld=896) ----
  gemm(attr_bf, FEAT, ae1_wt, N_NODES, 512, FEAT, ae1_b, ae1_g, ae1_be, 1, 0,
       ae_out, DCAT, ae_bf, DCAT);
  gemm(ae_bf, DCAT, ae2_wt, N_NODES, 256, 512, ae2_b, ae2_g, ae2_be, 1, 0,
       ae_out + 512, DCAT, ae_bf + 512, DCAT);
  gemm(ae_bf + 512, DCAT, ae3_wt, N_NODES, 128, 256, ae3_b, ae3_g, ae3_be, 1, 0,
       ae_out + 768, DCAT, ae_bf + 768, DCAT);
  // ---- decoder -> re_out ----
  gemm(ae_bf + 768, DCAT, dw1t, N_NODES, 256, 128, db1, nullptr, nullptr, 1, 0,
       nullptr, 0, dec1, 256);
  gemm(dec1, 256, dw2t, N_NODES, 512, 256, db2, nullptr, nullptr, 1, 0,
       nullptr, 0, dec2, 512);
  gemm(dec2, 512, dw3t, N_NODES, 2048, 512, db3, nullptr, nullptr, 3, 0,
       re_out, 2048, nullptr, 0);

  // ---- SAGE stack (diff_edge_index), outputs packed into dg ----
  {
    const float* xf = ae_out; const bf16_t* xb = ae_bf;
    int colofs[3] = {0, 512, 768};
    for (int l = 0; l < 3; ++l) {
      int fin = fin_[l], fout = fout_[l];
      hipMemsetAsync(agg, 0, (size_t)N_NODES * fin * 4, stream);
      hipMemsetAsync(cnt, 0, (size_t)N_NODES * 4, stream);
      k_scatter_feats<<<NE, 256, 0, stream>>>(xf, DCAT, diff_ei, diff_ei + NE, nullptr, 1, agg, fin);
      k_scatter_counts<<<(NE + 255) / 256, 256, 0, stream>>>(diff_ei + NE, nullptr, 1, cnt, NE);
      size_t tot = (size_t)N_NODES * fin;
      k_seg_div_bf16<<<(int)((tot + 255) / 256), 256, 0, stream>>>(agg, cnt, agg_bf, tot, fin);
      gemm(xb, DCAT, s_wrt[l], N_NODES, fout, fin, nullptr, nullptr, nullptr, 0, 0,
           dg + colofs[l], DCAT, nullptr, 0);                                    // x @ wr
      gemm(agg_bf, fin, s_wlt[l], N_NODES, fout, fin, s_bl[l], nullptr, nullptr, 2, 1,
           dg + colofs[l], DCAT, dg_bf + colofs[l], DCAT);                       // relu(+agg@wl+b)
      xf = dg + colofs[l]; xb = dg_bf + colofs[l];
    }
  }

  // ---- RGCN stack (edge_index, edge_type), einsum as one flat GEMM ----
  {
    const float* xf = ae_out; const bf16_t* xb = ae_bf;
    int colofs[3] = {0, 512, 768};
    for (int l = 0; l < 3; ++l) {
      int fin = fin_[l], fout = fout_[l];
      hipMemsetAsync(aggr, 0, (size_t)N_NODES * RELS * fin * 4, stream);
      hipMemsetAsync(cntr, 0, (size_t)N_NODES * RELS * 4, stream);
      k_scatter_feats<<<NE, 256, 0, stream>>>(xf, DCAT, edge_index, edge_index + NE, etype, RELS, aggr, fin);
      k_scatter_counts<<<(NE + 255) / 256, 256, 0, stream>>>(edge_index + NE, etype, RELS, cntr, NE);
      size_t tot = (size_t)N_NODES * RELS * fin;
      k_seg_div_bf16<<<(int)((tot + 255) / 256), 256, 0, stream>>>(aggr, cntr, aggr_bf, tot, fin);
      gemm(aggr_bf, RELS * fin, r_wrelt[l], N_NODES, fout, RELS * fin,
           nullptr, nullptr, nullptr, 0, 0, og + colofs[l], DCAT, nullptr, 0);   // einsum
      gemm(xb, DCAT, r_roott[l], N_NODES, fout, fin, r_bias[l], nullptr, nullptr, 1, 1,
           og + colofs[l], DCAT, og_bf + colofs[l], DCAT);                       // elu(+x@root+b)
      xf = og + colofs[l]; xb = og_bf + colofs[l];
    }
  }

  // ---- attention fuse ----
  k_att_scores<<<(N_NODES + 255) / 256, 256, 0, stream>>>(ae_out, dg, og, att_w1, att_b1, att_w2, beta);
  {
    size_t tot = (size_t)N_NODES * DCAT;
    k_att_combine<<<(int)((tot + 255) / 256), 256, 0, stream>>>(ae_out, dg, og, beta, emb);
  }

  // ---- triple pair DNN ----
  k_pair<<<NT, 256, 0, stream>>>(emb, triples, pairb);
  gemm(pairb, DCAT, d1_wt, NT, 512, DCAT, d1_b, d1_g, d1_be, 1, 0, nullptr, 0, h1, 512);
  gemm(h1, 512, d2_wt, NT, 256, 512, d2_b, d2_g, d2_be, 1, 0, nullptr, 0, h2, 256);
  gemm(h2, 256, d3_wt, NT, 8, 256, d3_b, nullptr, nullptr, 0, 0, out_fc, 8, nullptr, 0);
}
